// Attention_84748294685097
// MI455X (gfx1250) — matrix operand order repaired and checked by reading, not compiled, not run on hardware
//
#include <hip/hip_runtime.h>

// ---------------------------------------------------------------- types
typedef __bf16 bf16;
typedef __attribute__((ext_vector_type(16))) bf16  v16bf;
typedef __attribute__((ext_vector_type(8)))  float v8f;
typedef __attribute__((ext_vector_type(4)))  unsigned int u32x4; // trivial 16B vec
typedef __attribute__((ext_vector_type(4)))  float f32x4;

union FragBF {            // 32B WMMA operand (16 bf16) built from two 16B loads
    v16bf v;
    u32x4 q[2];
};

#define WMMA_BF16(a, b, c) \
    __builtin_amdgcn_wmma_f32_16x16x32_bf16(false, (a), false, (b), (short)0, (c), false, false)

static __device__ inline v8f v8f_zero() {
    v8f z;
    for (int i = 0; i < 8; ++i) z[i] = 0.0f;
    return z;
}

// Problem constants
#define BB 2
#define SS 2048
#define EE 1024
#define HH 16
#define DD 64

// =================================================================
// Kernel 1: QKV projection.  C[M=4096,N=1024] = X * W^T, W row-major (N,K).
// Block tile 128x128, 8 waves (4Mx2N), 32x64 per wave, K-step 32.
// Per K-step per wave: 8 WMMA vs 6 fragment loads (vs 4:4 before) ->
// much better matrix-pipe occupancy per unit of LDS bandwidth.
// blockIdx.z selects {Wq,Wk,Wv} -> {Qb,Kb,Vb} (bf16 outputs).
// =================================================================
__global__ __launch_bounds__(256)
void gemm_qkv_kernel(const float* __restrict__ X,
                     const float* __restrict__ Wq,
                     const float* __restrict__ Wk,
                     const float* __restrict__ Wv,
                     bf16* __restrict__ Qb,
                     bf16* __restrict__ Kb,
                     bf16* __restrict__ Vb) {
    const float* W   = (blockIdx.z == 0) ? Wq : (blockIdx.z == 1) ? Wk : Wv;
    bf16*        Out = (blockIdx.z == 0) ? Qb : (blockIdx.z == 1) ? Kb : Vb;

    __shared__ bf16 As[128 * 32];
    __shared__ bf16 Bs[128 * 32];

    const int tid  = threadIdx.x;
    const int lane = tid & 31;
    const int wave = tid >> 5;
    const int wm   = wave >> 1;   // 0..3 (M)
    const int wn   = wave & 1;    // 0..1 (N), wave tile 32x64
    const int m0   = blockIdx.x * 128;
    const int n0   = blockIdx.y * 128;
    const int r    = lane & 15;
    const int kh   = lane >> 4;

    v8f acc[2][4];
    for (int i = 0; i < 2; ++i)
        for (int j = 0; j < 4; ++j) acc[i][j] = v8f_zero();

    for (int k0 = 0; k0 < EE; k0 += 32) {
        __syncthreads();
        // stage A tile 128x32 (f32 -> bf16)
        for (int p = 0; p < 4; ++p) {
            int v   = tid + p * 256;        // 0..1023, one f32x4 each
            int row = v >> 3;
            int c4  = (v & 7) * 4;
            f32x4 f = *(const f32x4*)(X + (size_t)(m0 + row) * EE + k0 + c4);
            As[row * 32 + c4 + 0] = (bf16)f.x;
            As[row * 32 + c4 + 1] = (bf16)f.y;
            As[row * 32 + c4 + 2] = (bf16)f.z;
            As[row * 32 + c4 + 3] = (bf16)f.w;
        }
        // stage B tile 128x32 (W rows n0..n0+127, f32 -> bf16)
        for (int p = 0; p < 4; ++p) {
            int v   = tid + p * 256;        // 0..1023
            int row = v >> 3;
            int c4  = (v & 7) * 4;
            f32x4 f = *(const f32x4*)(W + (size_t)(n0 + row) * EE + k0 + c4);
            Bs[row * 32 + c4 + 0] = (bf16)f.x;
            Bs[row * 32 + c4 + 1] = (bf16)f.y;
            Bs[row * 32 + c4 + 2] = (bf16)f.z;
            Bs[row * 32 + c4 + 3] = (bf16)f.w;
        }
        __syncthreads();

        if (k0 + 32 < EE)  // pull next weight tile toward L2/WGP$
            __builtin_prefetch(W + (size_t)(n0 + (tid >> 1)) * EE + k0 + 32, 0, 1);

        FragBF a[2], b[4];
        for (int i = 0; i < 2; ++i) {
            const bf16* base = &As[(wm * 32 + i * 16 + r) * 32 + kh * 8];
            a[i].q[0] = *(const u32x4*)(base);
            a[i].q[1] = *(const u32x4*)(base + 16);
        }
        for (int j = 0; j < 4; ++j) {
            const bf16* base = &Bs[(wn * 64 + j * 16 + r) * 32 + kh * 8];
            b[j].q[0] = *(const u32x4*)(base);
            b[j].q[1] = *(const u32x4*)(base + 16);
        }
        for (int i = 0; i < 2; ++i)
            for (int j = 0; j < 4; ++j)
                acc[i][j] = WMMA_BF16(a[i].v, b[j].v, acc[i][j]);
    }

    const int half = kh * 8;
    for (int i = 0; i < 2; ++i)
        for (int j = 0; j < 4; ++j)
            for (int e = 0; e < 8; ++e) {
                int m  = m0 + wm * 32 + i * 16 + e + half;
                int nn = n0 + wn * 64 + j * 16 + r;
                Out[(size_t)m * EE + nn] = (bf16)acc[i][j][e];
            }
}

// =================================================================
// Kernel 2: RoPE in place on Q,K (bf16), interleaved pairs, f32 math.
// One thread per (row, pair).
// =================================================================
__global__ __launch_bounds__(256)
void rope_kernel(bf16* __restrict__ Q, bf16* __restrict__ K,
                 const float* __restrict__ sine, const float* __restrict__ cose) {
    int idx = blockIdx.x * 256 + threadIdx.x;   // 0 .. B*S*(E/2)-1
    int p   = idx & 511;                        // pair index within row
    int row = idx >> 9;                         // b*S + s
    int s   = row & (SS - 1);
    int d2  = p & (DD / 2 - 1);                 // frequency index 0..31
    float sn = sine[s * (DD / 2) + d2];
    float cs = cose[s * (DD / 2) + d2];
    size_t o = (size_t)row * EE + p * 2;
    float q1 = (float)Q[o], q2 = (float)Q[o + 1];
    Q[o]     = (bf16)(q1 * cs - q2 * sn);
    Q[o + 1] = (bf16)(q1 * sn + q2 * cs);
    float k1 = (float)K[o], k2 = (float)K[o + 1];
    K[o]     = (bf16)(k1 * cs - k2 * sn);
    K[o + 1] = (bf16)(k1 * sn + k2 * cs);
}

// =================================================================
// Kernel 3: causal flash attention.  Block = (128 q-rows, one (b,h)).
// 8 waves, 16 q-rows/wave.  Keys in blocks of 32 staged in LDS
// (K row-major, V transposed), softmax online in f32, P->LDS->A-frag.
// =================================================================
__global__ __launch_bounds__(256)
void attention_kernel(const bf16* __restrict__ Q, const bf16* __restrict__ Kg,
                      const bf16* __restrict__ Vg, bf16* __restrict__ Og) {
    __shared__ bf16 Ks[32 * 64];        // [key][d]
    __shared__ bf16 Vt[64 * 32];        // [d][key]
    __shared__ bf16 Ps[8 * 16 * 32];    // per-wave P tile [row][key]

    const int tid  = threadIdx.x;
    const int lane = tid & 31;
    const int wave = tid >> 5;
    const int q0   = blockIdx.x * 128;
    const int bh   = blockIdx.y;        // b*H + h
    const int b    = bh >> 4;
    const int h    = bh & 15;
    const size_t base = ((size_t)b * SS) * EE + (size_t)h * DD;

    const int r  = lane & 15;
    const int kh = lane >> 4;
    const int myq  = q0 + wave * 16;
    const int qrow = myq + r;

    // preload Q A-fragments for both 32-wide d-chunks
    FragBF qa[2];
    for (int c = 0; c < 2; ++c) {
        const bf16* p0 = Q + base + (size_t)qrow * EE + c * 32 + kh * 8;
        qa[c].q[0] = *(const u32x4*)(p0);
        qa[c].q[1] = *(const u32x4*)(p0 + 16);
    }

    v8f  o[4];
    float m[8], l[8];
    for (int dt = 0; dt < 4; ++dt) o[dt] = v8f_zero();
    for (int e = 0; e < 8; ++e) { m[e] = -1e30f; l[e] = 0.0f; }

    const int kend = q0 + 128;          // causal upper bound for this block
    for (int kb = 0; kb < kend; kb += 32) {
        __syncthreads();
        {   // cooperative load: 32 keys x 64 dims, K straight, V transposed
            int rowk = tid >> 3;        // 0..31
            int dst  = (tid & 7) * 8;   // 0..56
            *(u32x4*)&Ks[rowk * 64 + dst] =
                *(const u32x4*)(Kg + base + (size_t)(kb + rowk) * EE + dst);
            u32x4 vv = *(const u32x4*)(Vg + base + (size_t)(kb + rowk) * EE + dst);
            const bf16* ve = (const bf16*)&vv;
            for (int i = 0; i < 8; ++i) Vt[(dst + i) * 32 + rowk] = ve[i];
        }
        __syncthreads();

        // S = Q * K^T : two 16x16 C tiles covering 32 keys
        v8f s[2];
        for (int t = 0; t < 2; ++t) {
            s[t] = v8f_zero();
            for (int c = 0; c < 2; ++c) {
                FragBF kf;
                const bf16* p0 = &Ks[(t * 16 + r) * 64 + c * 32 + kh * 8];
                kf.q[0] = *(const u32x4*)p0;
                kf.q[1] = *(const u32x4*)(p0 + 16);
                s[t] = WMMA_BF16(qa[c].v, kf.v, s[t]);
            }
        }
        // scale + causal mask (lane holds key column, VGPR e holds row)
        for (int t = 0; t < 2; ++t) {
            int key = kb + t * 16 + r;
            for (int e = 0; e < 8; ++e) {
                int qq = myq + e + kh * 8;
                float val = s[t][e] * 0.125f;     // 1/sqrt(64)
                s[t][e] = (key <= qq) ? val : -1e30f;
            }
        }
        // online softmax: 16-lane butterfly reductions (wave32 halves)
        float alpha[8];
        for (int e = 0; e < 8; ++e) {
            float v = fmaxf(s[0][e], s[1][e]);
            v = fmaxf(v, __shfl_xor(v, 1, 32));
            v = fmaxf(v, __shfl_xor(v, 2, 32));
            v = fmaxf(v, __shfl_xor(v, 4, 32));
            v = fmaxf(v, __shfl_xor(v, 8, 32));
            float mn = fmaxf(m[e], v);
            alpha[e] = __expf(m[e] - mn);
            m[e] = mn;
        }
        for (int e = 0; e < 8; ++e) {
            float p0 = __expf(s[0][e] - m[e]);
            float p1 = __expf(s[1][e] - m[e]);
            s[0][e] = p0; s[1][e] = p1;
            float v = p0 + p1;
            v += __shfl_xor(v, 1, 32);
            v += __shfl_xor(v, 2, 32);
            v += __shfl_xor(v, 4, 32);
            v += __shfl_xor(v, 8, 32);
            l[e] = l[e] * alpha[e] + v;
        }
        for (int dt = 0; dt < 4; ++dt)
            for (int e = 0; e < 8; ++e) o[dt][e] *= alpha[e];

        // C-layout P -> per-wave LDS -> A-layout fragment
        bf16* pw = &Ps[wave * 512];
        for (int t = 0; t < 2; ++t)
            for (int e = 0; e < 8; ++e)
                pw[(e + kh * 8) * 32 + t * 16 + r] = (bf16)s[t][e];
        asm volatile("s_wait_dscnt 0" ::: "memory");   // CDNA5 split DS counter
        FragBF pa;
        {
            const bf16* p0 = &pw[r * 32 + kh * 8];
            pa.q[0] = *(const u32x4*)p0;
            pa.q[1] = *(const u32x4*)(p0 + 16);
        }
        // O += P * V  (B-frags from transposed V: two 16B ds reads, 32B apart)
        for (int dt = 0; dt < 4; ++dt) {
            FragBF vb;
            const bf16* p0 = &Vt[(dt * 16 + r) * 32 + kh * 8];
            vb.q[0] = *(const u32x4*)p0;
            vb.q[1] = *(const u32x4*)(p0 + 16);
            o[dt] = WMMA_BF16(pa.v, vb.v, o[dt]);
        }
    }

    // normalize + store bf16 attention output
    for (int e = 0; e < 8; ++e) {
        float inv  = 1.0f / l[e];
        int   mrow = myq + e + kh * 8;
        for (int dt = 0; dt < 4; ++dt)
            Og[base + (size_t)mrow * EE + dt * 16 + r] = (bf16)(o[dt][e] * inv);
    }
}

// =================================================================
// Kernel 4: output projection.  out(f32) = attn(bf16) @ Wo^T(f32->bf16).
// Same 128x128 tile engine as kernel 1 with a bf16 A path.
// =================================================================
__global__ __launch_bounds__(256)
void gemm_out_kernel(const bf16* __restrict__ A, const float* __restrict__ Wo,
                     float* __restrict__ Out) {
    __shared__ bf16 As[128 * 32];
    __shared__ bf16 Bs[128 * 32];

    const int tid  = threadIdx.x;
    const int lane = tid & 31;
    const int wave = tid >> 5;
    const int wm   = wave >> 1;
    const int wn   = wave & 1;
    const int m0   = blockIdx.x * 128;
    const int n0   = blockIdx.y * 128;
    const int r    = lane & 15;
    const int kh   = lane >> 4;

    v8f acc[2][4];
    for (int i = 0; i < 2; ++i)
        for (int j = 0; j < 4; ++j) acc[i][j] = v8f_zero();

    for (int k0 = 0; k0 < EE; k0 += 32) {
        __syncthreads();
        for (int p = 0; p < 2; ++p) {       // A tile: 4096 bf16 via 16B vectors
            int v   = tid + p * 256;        // 0..511
            int row = v >> 2;
            int c8  = (v & 3) * 8;
            *(u32x4*)&As[row * 32 + c8] =
                *(const u32x4*)(A + (size_t)(m0 + row) * EE + k0 + c8);
        }
        for (int p = 0; p < 4; ++p) {       // Wo tile 128x32 f32 -> bf16
            int v   = tid + p * 256;
            int row = v >> 3;
            int c4  = (v & 7) * 4;
            f32x4 f = *(const f32x4*)(Wo + (size_t)(n0 + row) * EE + k0 + c4);
            Bs[row * 32 + c4 + 0] = (bf16)f.x;
            Bs[row * 32 + c4 + 1] = (bf16)f.y;
            Bs[row * 32 + c4 + 2] = (bf16)f.z;
            Bs[row * 32 + c4 + 3] = (bf16)f.w;
        }
        __syncthreads();

        if (k0 + 32 < EE)
            __builtin_prefetch(Wo + (size_t)(n0 + (tid >> 1)) * EE + k0 + 32, 0, 1);

        FragBF a[2], b[4];
        for (int i = 0; i < 2; ++i) {
            const bf16* base = &As[(wm * 32 + i * 16 + r) * 32 + kh * 8];
            a[i].q[0] = *(const u32x4*)(base);
            a[i].q[1] = *(const u32x4*)(base + 16);
        }
        for (int j = 0; j < 4; ++j) {
            const bf16* base = &Bs[(wn * 64 + j * 16 + r) * 32 + kh * 8];
            b[j].q[0] = *(const u32x4*)(base);
            b[j].q[1] = *(const u32x4*)(base + 16);
        }
        for (int i = 0; i < 2; ++i)
            for (int j = 0; j < 4; ++j)
                acc[i][j] = WMMA_BF16(a[i].v, b[j].v, acc[i][j]);
    }

    const int half = kh * 8;
    for (int i = 0; i < 2; ++i)
        for (int j = 0; j < 4; ++j)
            for (int e = 0; e < 8; ++e) {
                int m  = m0 + wm * 32 + i * 16 + e + half;
                int nn = n0 + wn * 64 + j * 16 + r;
                Out[(size_t)m * EE + nn] = acc[i][j][e];
            }
}

// ---------------------------------------------------------------- launch
extern "C" void kernel_launch(void* const* d_in, const int* in_sizes, int n_in,
                              void* d_out, int out_size, void* d_ws, size_t ws_size,
                              hipStream_t stream) {
    (void)in_sizes; (void)n_in; (void)out_size; (void)ws_size;
    // setup_inputs order: x, sin_emb, cos_emb, mask, Wq, Wk, Wv, Wo
    const float* x    = (const float*)d_in[0];
    const float* sine = (const float*)d_in[1];
    const float* cose = (const float*)d_in[2];
    const float* Wq   = (const float*)d_in[4];
    const float* Wk   = (const float*)d_in[5];
    const float* Wv   = (const float*)d_in[6];
    const float* Wo   = (const float*)d_in[7];
    float* out = (float*)d_out;

    const size_t mat = (size_t)BB * SS * EE;   // 4096*1024 elements
    bf16* Qb = (bf16*)d_ws;
    bf16* Kb = Qb + mat;
    bf16* Vb = Kb + mat;
    bf16* Ob = Vb + mat;                       // attention output (bf16)

    // 1) QKV projections (z selects weight/output)
    gemm_qkv_kernel<<<dim3(32, 8, 3), 256, 0, stream>>>(x, Wq, Wk, Wv, Qb, Kb, Vb);
    // 2) RoPE on Q,K in place
    rope_kernel<<<dim3((BB * SS * (EE / 2)) / 256), 256, 0, stream>>>(Qb, Kb, sine, cose);
    // 3) causal flash attention per (b,h,128-row q tile)
    attention_kernel<<<dim3(SS / 128, BB * HH), 256, 0, stream>>>(Qb, Kb, Vb, Ob);
    // 4) output projection -> f32
    gemm_out_kernel<<<dim3(32, 8, 1), 256, 0, stream>>>(Ob, Wo, out);
}
